// EncHead_83494164234636
// MI455X (gfx1250) — matrix operand order repaired and checked by reading, not compiled, not run on hardware
//
#include <hip/hip_runtime.h>

// ============================================================================
// EncHead for MI455X (gfx1250, wave32, WMMA + async-to-LDS staging).
// All channel contractions (1x1 convs, 3x3 conv as 9 implicit-GEMM taps,
// codeword assignment, E-aggregation, classifier) run through
// v_wmma_f32_16x16x32_f16. B tiles are staged to LDS with
// global_load_async_to_lds_b32 (ASYNCcnt) and read back as ds_load_b128.
// ============================================================================

typedef __attribute__((ext_vector_type(16))) _Float16 v16h;
typedef __attribute__((ext_vector_type(8)))  float    v8f;

#define BN_EPS 1e-5f
#define LDS_PITCH 40   // floats per LDS row: 16B-aligned rows, bank-spread

// ---------------------------------------------------------------------------
// Generic WMMA GEMM:  C[b][m][n] (+)= sum_k A[m][k]*B[b][k][n] (+ bias[m])
//   A index: b*aB + aoff + m*ldaRow + k*ldaK
//   B index: b*bB + { mode0: k*ldbK + n
//                     mode1: k*ldbK + (n/bW)*bPW + n%bW + boff   (padded 3x3)
//                     mode2: n*ldbK + k }                        (B transposed)
// Block = 128 threads = 4 waves. Block tile 16M x 128N; each wave computes
// two 16x16 C tiles sharing one A fragment.
// ---------------------------------------------------------------------------
struct GemmP {
  const float* A; const float* B; float* C; const float* bias;
  int M, N, K;
  long aB, bB, cB;
  int ldaRow, ldaK, aoff;
  int bmode, ldbK, bW, bPW, boff;
  int acc;
};

__device__ __forceinline__ v16h load_bfrag(const float* bp) {
  // 2x 8 contiguous f32 from LDS at bp and bp+16 (4x ds_load_b128) -> f16 fragment
  const float4 b0 = *(const float4*)(bp);
  const float4 b1 = *(const float4*)(bp + 4);
  const float4 b2 = *(const float4*)(bp + 16);
  const float4 b3 = *(const float4*)(bp + 20);
  v16h r;
  r[0]  = (_Float16)b0.x; r[1]  = (_Float16)b0.y;
  r[2]  = (_Float16)b0.z; r[3]  = (_Float16)b0.w;
  r[4]  = (_Float16)b1.x; r[5]  = (_Float16)b1.y;
  r[6]  = (_Float16)b1.z; r[7]  = (_Float16)b1.w;
  r[8]  = (_Float16)b2.x; r[9]  = (_Float16)b2.y;
  r[10] = (_Float16)b2.z; r[11] = (_Float16)b2.w;
  r[12] = (_Float16)b3.x; r[13] = (_Float16)b3.y;
  r[14] = (_Float16)b3.z; r[15] = (_Float16)b3.w;
  return r;
}

template <int BMODE, bool AK1>
__global__ __launch_bounds__(128)
void wmma_gemm_kernel(GemmP p) {
  __shared__ float ldsB[128 * LDS_PITCH];   // 20.5 KB B tile: [n][k] f32

  const int tid  = threadIdx.x;
  const int wave = tid >> 5;
  const int lane = tid & 31;
  const int hf   = lane >> 4;      // lane half 0|1
  const int l15  = lane & 15;
  const int b    = blockIdx.z;
  const int mtile  = blockIdx.y * 16;
  const int nblock = blockIdx.x * 128;

  const float* Ab = p.A + (long)b * p.aB + p.aoff;
  const float* Bb = p.B + (long)b * p.bB;
  float*       Cb = p.C + (long)b * p.cB;

  const int  m   = mtile + l15;
  const bool mok = (m < p.M);
  const float* arow = Ab + (long)(mok ? m : 0) * p.ldaRow;

  // per-thread B column offset for modes 0/1 (thread t loads global column
  // nblock+t for all 32 k of the current slab)
  long ncol = 0;
  if (BMODE == 1) {
    const int n = nblock + tid;
    ncol = (long)(n / p.bW) * p.bPW + (n % p.bW) + p.boff;
  } else if (BMODE == 0) {
    ncol = nblock + tid;
  }

  const unsigned ldsbase = (unsigned)(size_t)(&ldsB[0]);
  const float* bp0 = &ldsB[(wave * 32 + l15) * LDS_PITCH + hf * 8];
  const float* bp1 = bp0 + 16 * LDS_PITCH;

  v8f acc0 = {}, acc1 = {};

  for (int k0 = 0; k0 < p.K; k0 += 32) {
    // ---- async-stage the 32(K) x 128(N) B slab into LDS (f32, [n][k]) ----
    #pragma unroll
    for (int r = 0; r < 32; ++r) {
      const float* ga;
      unsigned ldst;
      if (BMODE == 2) {               // B transposed: contiguous along k
        const int nl = (tid >> 5) * 32 + r;
        ga   = Bb + (long)(nblock + nl) * p.ldbK + (k0 + (tid & 31));
        ldst = (unsigned)(nl * LDS_PITCH + (tid & 31));
      } else {                        // contiguous along n
        ga   = Bb + (long)(k0 + r) * p.ldbK + ncol;
        ldst = (unsigned)(tid * LDS_PITCH + r);
      }
      const unsigned dst = ldsbase + ldst * 4u;
      asm volatile("global_load_async_to_lds_b32 %0, %1, off"
                   :: "v"(dst), "v"(ga) : "memory");
    }
    asm volatile("s_wait_asynccnt 0x0" ::: "memory");
    __syncthreads();

    // ---- A fragment: 16x32 f16, ISA 7.12.2 layout ----
    __builtin_prefetch(arow + (long)(k0 + 32) * p.ldaK, 0, 1);
    v16h a = {};
    if (mok) {
      if (AK1) {
        const float4 r0 = *(const float4*)(arow + k0 + hf * 8);
        const float4 r1 = *(const float4*)(arow + k0 + hf * 8 + 4);
        const float4 r2 = *(const float4*)(arow + k0 + 16 + hf * 8);
        const float4 r3 = *(const float4*)(arow + k0 + 16 + hf * 8 + 4);
        a[0]  = (_Float16)r0.x; a[1]  = (_Float16)r0.y;
        a[2]  = (_Float16)r0.z; a[3]  = (_Float16)r0.w;
        a[4]  = (_Float16)r1.x; a[5]  = (_Float16)r1.y;
        a[6]  = (_Float16)r1.z; a[7]  = (_Float16)r1.w;
        a[8]  = (_Float16)r2.x; a[9]  = (_Float16)r2.y;
        a[10] = (_Float16)r2.z; a[11] = (_Float16)r2.w;
        a[12] = (_Float16)r3.x; a[13] = (_Float16)r3.y;
        a[14] = (_Float16)r3.z; a[15] = (_Float16)r3.w;
      } else {
        #pragma unroll
        for (int j = 0; j < 8; ++j) {
          a[j]     = (_Float16)arow[(long)(k0 + hf * 8 + j)      * p.ldaK];
          a[j + 8] = (_Float16)arow[(long)(k0 + 16 + hf * 8 + j) * p.ldaK];
        }
      }
    }

    // ---- B fragments from LDS, two N-subtiles per wave, 2x WMMA ----
    const v16h bv0 = load_bfrag(bp0);
    const v16h bv1 = load_bfrag(bp1);
    acc0 = __builtin_amdgcn_wmma_f32_16x16x32_f16(false, a, false, bv0,
                                                  (short)0, acc0, false, false);
    acc1 = __builtin_amdgcn_wmma_f32_16x16x32_f16(false, a, false, bv1,
                                                  (short)0, acc1, false, false);
    __syncthreads();   // protect single-buffered LDS tile
  }

  // ---- store: C/D layout VGPR i -> row i+hf*8, col l15 ----
  #pragma unroll
  for (int i = 0; i < 8; ++i) {
    const int mr = mtile + i + hf * 8;
    if (mr < p.M) {
      const long ci = (long)mr * p.N + nblock + wave * 32 + l15;
      float v0 = acc0[i], v1 = acc1[i];
      if (p.bias) { v0 += p.bias[mr]; v1 += p.bias[mr]; }
      if (p.acc)  { v0 += Cb[ci];     v1 += Cb[ci + 16]; }
      Cb[ci]      = v0;
      Cb[ci + 16] = v1;
    }
  }
}

// ---------------------------------------------------------------------------
// Bilinear upsample, align_corners=True.  grid = (ceil(oh*ow/256), C, B)
// ---------------------------------------------------------------------------
__global__ void bilinear_kernel(const float* in, float* out,
                                int ih, int iw, int oh, int ow) {
  const int p = blockIdx.x * blockDim.x + threadIdx.x;
  if (p >= oh * ow) return;
  const long c = (long)blockIdx.z * gridDim.y + blockIdx.y;
  const int oy = p / ow, ox = p % ow;
  const float fy = (oh > 1) ? (float)oy * (float)(ih - 1) / (float)(oh - 1) : 0.f;
  const float fx = (ow > 1) ? (float)ox * (float)(iw - 1) / (float)(ow - 1) : 0.f;
  int y0 = (int)floorf(fy); int y1 = y0 + 1; if (y1 > ih - 1) y1 = ih - 1;
  int x0 = (int)floorf(fx); int x1 = x0 + 1; if (x1 > iw - 1) x1 = iw - 1;
  const float wy = fy - (float)y0, wx = fx - (float)x0;
  const float* ip = in + c * (long)(ih * iw);
  const float top = ip[y0 * iw + x0] * (1.f - wx) + ip[y0 * iw + x1] * wx;
  const float bot = ip[y1 * iw + x0] * (1.f - wx) + ip[y1 * iw + x1] * wx;
  out[c * (long)(oh * ow) + p] = top * (1.f - wy) + bot * wy;
}

// ---------------------------------------------------------------------------
// Training-mode BN statistics per channel over (B, P).  grid = (C), 256 thr.
// ---------------------------------------------------------------------------
__global__ void bn_stats_kernel(const float* x, float* mean, float* istd,
                                int B, int C, long P) {
  const int c = blockIdx.x;
  float s = 0.f, s2 = 0.f;
  for (int b = 0; b < B; ++b) {
    const float* xp = x + ((long)b * C + c) * P;
    for (long p = threadIdx.x; p < P; p += blockDim.x) {
      const float v = xp[p]; s += v; s2 += v * v;
    }
  }
  __shared__ float sh0[256], sh1[256];
  sh0[threadIdx.x] = s; sh1[threadIdx.x] = s2;
  __syncthreads();
  for (int o = 128; o > 0; o >>= 1) {
    if ((int)threadIdx.x < o) {
      sh0[threadIdx.x] += sh0[threadIdx.x + o];
      sh1[threadIdx.x] += sh1[threadIdx.x + o];
    }
    __syncthreads();
  }
  if (threadIdx.x == 0) {
    const float n = (float)B * (float)P;
    const float m = sh0[0] / n;
    const float v = sh1[0] / n - m * m;
    mean[c] = m;
    istd[c] = rsqrtf(v + BN_EPS);
  }
}

// grid = (ceil(P/256), C, B)
__global__ void bn_apply_kernel(float* x, const float* mean, const float* istd,
                                const float* g, const float* bt,
                                int C, long P, int relu) {
  const long p = (long)blockIdx.x * blockDim.x + threadIdx.x;
  if (p >= P) return;
  const int c = blockIdx.y;
  const long idx = ((long)blockIdx.z * C + c) * P + p;
  float v = (x[idx] - mean[c]) * istd[c] * g[c] + bt[c];
  if (relu) v = fmaxf(v, 0.f);
  x[idx] = v;
}

// ---------------------------------------------------------------------------
// Zero-pad (pad=1) NCHW -> NC(H+2)(W+2).  grid = (ceil(PH*PW/256), C, B)
// ---------------------------------------------------------------------------
__global__ void pad_kernel(const float* in, float* out, int C, int H, int W) {
  const int PH = H + 2, PW = W + 2;
  const int p = blockIdx.x * blockDim.x + threadIdx.x;
  if (p >= PH * PW) return;
  const long c = (long)blockIdx.z * C + blockIdx.y;
  const int y = p / PW, x = p % PW;
  float v = 0.f;
  if (y >= 1 && y <= H && x >= 1 && x <= W)
    v = in[c * (long)(H * W) + (long)(y - 1) * W + (x - 1)];
  out[c * (long)(PH * PW) + p] = v;
}

// ---------------------------------------------------------------------------
// 9-tap dilated (d=2, pad=2) attention: energy + softmax.
// q,k: (B,kd,P) post-BN.  att out: (B,9,P).  grid = (ceil(P/256), B)
// ---------------------------------------------------------------------------
__global__ void attn_softmax_kernel(const float* q, const float* k, float* att,
                                    int kd, int H, int W) {
  const int P = H * W;
  const int p = blockIdx.x * blockDim.x + threadIdx.x;
  if (p >= P) return;
  const int b = blockIdx.y;
  const int h = p / W, w = p % W;
  const float* qb = q + (long)b * kd * P;
  const float* kb = k + (long)b * kd * P;
  float e[9];
  int t = 0;
  for (int dy = -2; dy <= 2; dy += 2)
    for (int dx = -2; dx <= 2; dx += 2, ++t) {
      const int hy = h + dy, wx = w + dx;
      float s = 0.f;
      if (hy >= 0 && hy < H && wx >= 0 && wx < W) {
        const int pn = hy * W + wx;
        for (int c = 0; c < kd; ++c)
          s += qb[(long)c * P + p] * kb[(long)c * P + pn];
      }
      e[t] = s;                     // zero-padded unfold => energy 0, not -inf
    }
  float mx = e[0];
  for (int i = 1; i < 9; ++i) mx = fmaxf(mx, e[i]);
  float sum = 0.f;
  for (int i = 0; i < 9; ++i) { e[i] = __expf(e[i] - mx); sum += e[i]; }
  const float inv = 1.f / sum;
  for (int i = 0; i < 9; ++i) att[((long)b * 9 + i) * P + p] = e[i] * inv;
}

// out[c,p] = sum_t att[t,p] * v[c, neighbor_t(p)].  grid=(ceil(P/256), C1, B)
__global__ void attn_apply_kernel(const float* att, const float* v, float* out,
                                  int C1, int H, int W) {
  const int P = H * W;
  const int p = blockIdx.x * blockDim.x + threadIdx.x;
  if (p >= P) return;
  const int c = blockIdx.y, b = blockIdx.z;
  const int h = p / W, w = p % W;
  const float* ab = att + (long)b * 9 * P;
  const float* vb = v + ((long)b * C1 + c) * P;
  float s = 0.f; int t = 0;
  for (int dy = -2; dy <= 2; dy += 2)
    for (int dx = -2; dx <= 2; dx += 2, ++t) {
      const int hy = h + dy, wx = w + dx;
      if (hy >= 0 && hy < H && wx >= 0 && wx < W)
        s += ab[(long)t * P + p] * vb[hy * W + wx];
    }
  out[((long)b * C1 + c) * P + p] = s;
}

// ---------------------------------------------------------------------------
// Encoding-module helpers
// ---------------------------------------------------------------------------
__global__ void sqnorm_kernel(const float* x, float* xx, int D, long P) {
  const long p = (long)blockIdx.x * blockDim.x + threadIdx.x;
  if (p >= P) return;
  const float* xb = x + (long)blockIdx.y * D * P;
  float s = 0.f;
  for (int d = 0; d < D; ++d) { const float v = xb[(long)d * P + p]; s += v * v; }
  xx[(long)blockIdx.y * P + p] = s;
}

__global__ void cwnorm_kernel(const float* cw, float* cc, int K, int D) {
  const int k = blockIdx.x * blockDim.x + threadIdx.x;
  if (k >= K) return;
  float s = 0.f;
  for (int d = 0; d < D; ++d) { const float v = cw[k * D + d]; s += v * v; }
  cc[k] = s;
}

// A = softmax_k(-scale_k*(xx + cc_k - 2*xc_k)), stored (B,K,P). grid=(ceil(P/256),B)
__global__ void assign_kernel(const float* xc, const float* xx, const float* cc,
                              const float* scale, float* Aout, long P) {
  const long p = (long)blockIdx.x * blockDim.x + threadIdx.x;
  if (p >= P) return;
  const int b = blockIdx.y;
  const float xv = xx[(long)b * P + p];
  float e[32];
  float mx = -1e30f;
  for (int k = 0; k < 32; ++k) {
    const float d2 = xv + cc[k] - 2.f * xc[((long)b * 32 + k) * P + p];
    e[k] = -scale[k] * d2;
    mx = fmaxf(mx, e[k]);
  }
  float sum = 0.f;
  for (int k = 0; k < 32; ++k) { e[k] = __expf(e[k] - mx); sum += e[k]; }
  const float inv = 1.f / sum;
  for (int k = 0; k < 32; ++k) Aout[((long)b * 32 + k) * P + p] = e[k] * inv;
}

// row sums of A: grid = (B*K), 256 threads
__global__ void rowsum_kernel(const float* A, float* s, long P) {
  const float* row = A + (long)blockIdx.x * P;
  float acc = 0.f;
  for (long p = threadIdx.x; p < P; p += blockDim.x) acc += row[p];
  __shared__ float sh[256];
  sh[threadIdx.x] = acc; __syncthreads();
  for (int o = 128; o > 0; o >>= 1) {
    if ((int)threadIdx.x < o) sh[threadIdx.x] += sh[threadIdx.x + o];
    __syncthreads();
  }
  if (threadIdx.x == 0) s[blockIdx.x] = sh[0];
}

__global__ void efix_kernel(float* E, const float* asum, const float* cw,
                            int B, int K, int D) {
  const int i = blockIdx.x * blockDim.x + threadIdx.x;
  if (i >= B * K * D) return;
  const int d = i % D, bk = i / D, k = bk % K;
  E[i] -= asum[bk] * cw[k * D + d];
}

__global__ void enmean_kernel(const float* E, float* en, int B, int K, int D) {
  const int i = blockIdx.x * blockDim.x + threadIdx.x;
  if (i >= B * D) return;
  const int b = i / D, d = i % D;
  float s = 0.f;
  for (int k = 0; k < K; ++k) s += E[((long)b * K + k) * D + d];
  en[i] = s / (float)K;
}

__global__ void fc_kernel(const float* en, const float* Wt, const float* bias,
                          float* out, int B, int D, int O, int sigm) {
  const int i = blockIdx.x * blockDim.x + threadIdx.x;
  if (i >= B * O) return;
  const int b = i / O, o = i % O;
  float s = bias[o];
  for (int d = 0; d < D; ++d) s += en[b * D + d] * Wt[o * D + d];
  if (sigm) s = 1.f / (1.f + __expf(-s));
  out[i] = s;
}

__global__ void scale_res_kernel(const float* feat, const float* gamma,
                                 float* y, int C, long P) {
  const long p = (long)blockIdx.x * blockDim.x + threadIdx.x;
  if (p >= P) return;
  const int c = blockIdx.y, b = blockIdx.z;
  const long idx = ((long)b * C + c) * P + p;
  const float f = feat[idx];
  y[idx] = fmaxf(f + f * gamma[b * C + c], 0.f);
}

// ---------------------------------------------------------------------------
// Host orchestration
// ---------------------------------------------------------------------------
static void run_gemm(hipStream_t s, const float* A, const float* B, float* C,
                     const float* bias, int M, int N, int K,
                     long aB, long bB, long cB,
                     int ldaRow, int ldaK, int aoff,
                     int bmode, int ldbK, int bW, int bPW, int boff,
                     int acc, int batch) {
  GemmP p{A, B, C, bias, M, N, K, aB, bB, cB,
          ldaRow, ldaK, aoff, bmode, ldbK, bW, bPW, boff, acc};
  dim3 g((N + 127) / 128, (M + 15) / 16, batch);
  if (bmode == 1)      wmma_gemm_kernel<1, false><<<g, dim3(128), 0, s>>>(p);
  else if (bmode == 2) wmma_gemm_kernel<2, true ><<<g, dim3(128), 0, s>>>(p);
  else                 wmma_gemm_kernel<0, true ><<<g, dim3(128), 0, s>>>(p);
}

extern "C" void kernel_launch(void* const* d_in, const int* in_sizes, int n_in,
                              void* d_out, int out_size, void* d_ws, size_t ws_size,
                              hipStream_t stream) {
  (void)in_sizes; (void)n_in; (void)out_size; (void)ws_size;
  const float* c2  = (const float*)d_in[1];   // (2,512,64,64)
  const float* c3  = (const float*)d_in[2];   // (2,1024,32,32)
  const float* c4  = (const float*)d_in[3];   // (2,2048,16,16)
  const float* wq4 = (const float*)d_in[7];
  const float* gq4 = (const float*)d_in[8];
  const float* bq4 = (const float*)d_in[9];
  const float* wk4 = (const float*)d_in[10];
  const float* gk4 = (const float*)d_in[11];
  const float* bk4 = (const float*)d_in[12];
  const float* wv4 = (const float*)d_in[13];
  const float* wq3 = (const float*)d_in[14];
  const float* gq3 = (const float*)d_in[15];
  const float* bq3 = (const float*)d_in[16];
  const float* wk3 = (const float*)d_in[17];
  const float* gk3 = (const float*)d_in[18];
  const float* bk3 = (const float*)d_in[19];
  const float* wv3 = (const float*)d_in[20];
  const float* w5  = (const float*)d_in[21];  // (512,512,3,3)
  const float* g5  = (const float*)d_in[22];
  const float* b5  = (const float*)d_in[23];
  const float* we  = (const float*)d_in[24];
  const float* ge  = (const float*)d_in[25];
  const float* be  = (const float*)d_in[26];
  const float* cw  = (const float*)d_in[27];  // (32,512)
  const float* scl = (const float*)d_in[28];
  const float* gK  = (const float*)d_in[29];
  const float* bK  = (const float*)d_in[30];
  const float* wfc = (const float*)d_in[31];
  const float* bfc = (const float*)d_in[32];
  const float* wse = (const float*)d_in[33];
  const float* bse = (const float*)d_in[34];
  const float* w6  = (const float*)d_in[35];  // (150,512)
  const float* b6  = (const float*)d_in[36];
  float* out = (float*)d_out;

  const int Bn = 2;
  const long P4 = 1024, P3 = 4096;

  // ---- workspace bump allocator (floats), with aliasing reuse ----
  float* ws = (float*)d_ws;
  size_t off = 0;
  auto alloc = [&](size_t n) { float* p = ws + off; off += n; return p; };
  float* cku4  = alloc(2l * 2048 * P4);     // 16 MB (reused later as y)
  float* q4    = alloc(2l * 128 * P4);
  float* k4    = alloc(2l * 128 * P4);
  float* v4    = alloc(2l * 1024 * P4);
  float* att4  = alloc(2l * 9 * P4);
  float* out4  = alloc(2l * 1024 * P4);
  float* cku3  = alloc(2l * 1024 * P3);     // 32 MB (reused later as feat+x)
  float* q3    = alloc(2l * 64 * P3);
  float* k3    = alloc(2l * 64 * P3);
  float* v3    = alloc(2l * 512 * P3);
  float* att3  = alloc(2l * 9 * P3);
  float* out3  = alloc(2l * 512 * P3);
  float* out3p = alloc(2l * 512 * 66 * 66);
  float* xx    = alloc(2l * P3);
  float* ccn   = alloc(32);
  float* xc    = alloc(2l * 32 * P3);
  float* Am    = alloc(2l * 32 * P3);
  float* asum  = alloc(64);
  float* Ebuf  = alloc(2l * 32 * 512);
  float* en    = alloc(2l * 512);
  float* gamma = alloc(2l * 512);
  float* meanb = alloc(1024);
  float* istdb = alloc(1024);
  float* feat  = cku3;                      // alias: cku3 dead after v3 GEMM
  float* xbuf  = cku3 + 2l * 512 * P3;      // alias: second half of cku3
  float* ybuf  = cku4;                      // alias: cku4 dead after v4 GEMM

  // ================= local_up level 4: (c3, c4) -> out4 (2,1024,32,32) =====
  run_gemm(stream, wq4, c3, q4, nullptr, 128, 1024, 1024,
           0, 1024l * P4, 128l * P4, 1024, 1, 0, 0, (int)P4, 0, 0, 0, 0, Bn);
  bn_stats_kernel<<<dim3(128), 256, 0, stream>>>(q4, meanb, istdb, Bn, 128, P4);
  bn_apply_kernel<<<dim3(4, 128, Bn), 256, 0, stream>>>(q4, meanb, istdb, gq4, bq4, 128, P4, 0);

  bilinear_kernel<<<dim3(4, 2048, Bn), 256, 0, stream>>>(c4, cku4, 16, 16, 32, 32);

  run_gemm(stream, wk4, cku4, k4, nullptr, 128, 1024, 2048,
           0, 2048l * P4, 128l * P4, 2048, 1, 0, 0, (int)P4, 0, 0, 0, 0, Bn);
  bn_stats_kernel<<<dim3(128), 256, 0, stream>>>(k4, meanb, istdb, Bn, 128, P4);
  bn_apply_kernel<<<dim3(4, 128, Bn), 256, 0, stream>>>(k4, meanb, istdb, gk4, bk4, 128, P4, 0);

  run_gemm(stream, wv4, cku4, v4, nullptr, 1024, 1024, 2048,
           0, 2048l * P4, 1024l * P4, 2048, 1, 0, 0, (int)P4, 0, 0, 0, 0, Bn);

  attn_softmax_kernel<<<dim3(4, Bn), 256, 0, stream>>>(q4, k4, att4, 128, 32, 32);
  attn_apply_kernel<<<dim3(4, 1024, Bn), 256, 0, stream>>>(att4, v4, out4, 1024, 32, 32);

  // ================= local_up level 3: (c2, out4) -> out3 (2,512,64,64) ====
  run_gemm(stream, wq3, c2, q3, nullptr, 64, 4096, 512,
           0, 512l * P3, 64l * P3, 512, 1, 0, 0, (int)P3, 0, 0, 0, 0, Bn);
  bn_stats_kernel<<<dim3(64), 256, 0, stream>>>(q3, meanb, istdb, Bn, 64, P3);
  bn_apply_kernel<<<dim3(16, 64, Bn), 256, 0, stream>>>(q3, meanb, istdb, gq3, bq3, 64, P3, 0);

  bilinear_kernel<<<dim3(16, 1024, Bn), 256, 0, stream>>>(out4, cku3, 32, 32, 64, 64);

  run_gemm(stream, wk3, cku3, k3, nullptr, 64, 4096, 1024,
           0, 1024l * P3, 64l * P3, 1024, 1, 0, 0, (int)P3, 0, 0, 0, 0, Bn);
  bn_stats_kernel<<<dim3(64), 256, 0, stream>>>(k3, meanb, istdb, Bn, 64, P3);
  bn_apply_kernel<<<dim3(16, 64, Bn), 256, 0, stream>>>(k3, meanb, istdb, gk3, bk3, 64, P3, 0);

  run_gemm(stream, wv3, cku3, v3, nullptr, 512, 4096, 1024,
           0, 1024l * P3, 512l * P3, 1024, 1, 0, 0, (int)P3, 0, 0, 0, 0, Bn);

  attn_softmax_kernel<<<dim3(16, Bn), 256, 0, stream>>>(q3, k3, att3, 64, 64, 64);
  attn_apply_kernel<<<dim3(16, 512, Bn), 256, 0, stream>>>(att3, v3, out3, 512, 64, 64);

  // ================= conv5: 3x3 SAME as 9 accumulated implicit GEMMs =======
  pad_kernel<<<dim3(18, 512, Bn), 256, 0, stream>>>(out3, out3p, 512, 64, 64);
  for (int ty = 0; ty < 3; ++ty)
    for (int tx = 0; tx < 3; ++tx) {
      const int tap = ty * 3 + tx;
      run_gemm(stream, w5, out3p, feat, nullptr, 512, 4096, 512,
               0, 512l * 66 * 66, 512l * P3,
               512 * 9, 9, tap,                  // A: w5[o, i, ty, tx]
               1, 66 * 66, 64, 66, ty * 66 + tx, // B: padded im2col
               tap > 0, Bn);
    }
  bn_stats_kernel<<<dim3(512), 256, 0, stream>>>(feat, meanb, istdb, Bn, 512, P3);
  bn_apply_kernel<<<dim3(16, 512, Bn), 256, 0, stream>>>(feat, meanb, istdb, g5, b5, 512, P3, 1);

  // ================= encoding module =======================================
  run_gemm(stream, we, feat, xbuf, nullptr, 512, 4096, 512,
           0, 512l * P3, 512l * P3, 512, 1, 0, 0, (int)P3, 0, 0, 0, 0, Bn);
  bn_stats_kernel<<<dim3(512), 256, 0, stream>>>(xbuf, meanb, istdb, Bn, 512, P3);
  bn_apply_kernel<<<dim3(16, 512, Bn), 256, 0, stream>>>(xbuf, meanb, istdb, ge, be, 512, P3, 1);

  sqnorm_kernel<<<dim3(16, Bn), 256, 0, stream>>>(xbuf, xx, 512, P3);
  cwnorm_kernel<<<dim3(1), 32, 0, stream>>>(cw, ccn, 32, 512);

  run_gemm(stream, cw, xbuf, xc, nullptr, 32, 4096, 512,
           0, 512l * P3, 32l * P3, 512, 1, 0, 0, (int)P3, 0, 0, 0, 0, Bn);
  assign_kernel<<<dim3(16, Bn), 256, 0, stream>>>(xc, xx, ccn, scl, Am, P3);
  rowsum_kernel<<<dim3(64), 256, 0, stream>>>(Am, asum, P3);

  // E[b,k,d] = sum_p A[b,k,p]*x[b,d,p]  (B transposed mode), then subtract
  run_gemm(stream, Am, xbuf, Ebuf, nullptr, 32, 512, 4096,
           32l * P3, 512l * P3, 32l * 512, (int)P3, 1, 0, 2, (int)P3, 0, 0, 0, 0, Bn);
  efix_kernel<<<dim3(128), 256, 0, stream>>>(Ebuf, asum, cw, Bn, 32, 512);

  // BN1d over (b,d) per codeword channel k, + relu; then en = mean over k
  bn_stats_kernel<<<dim3(32), 256, 0, stream>>>(Ebuf, meanb, istdb, Bn, 32, 512);
  bn_apply_kernel<<<dim3(2, 32, Bn), 256, 0, stream>>>(Ebuf, meanb, istdb, gK, bK, 32, 512, 1);
  enmean_kernel<<<dim3(4), 256, 0, stream>>>(Ebuf, en, Bn, 32, 512);

  fc_kernel<<<dim3(4), 256, 0, stream>>>(en, wfc, bfc, gamma, Bn, 512, 512, 1);
  fc_kernel<<<dim3(2), 256, 0, stream>>>(en, wse, bse, out + 2l * 150 * P3, Bn, 512, 150, 0);

  scale_res_kernel<<<dim3(16, 512, Bn), 256, 0, stream>>>(feat, gamma, ybuf, 512, P3);

  // ================= classifier: seg = w6·y + b6 -> d_out ==================
  run_gemm(stream, w6, ybuf, out, b6, 150, 4096, 512,
           0, 512l * P3, 150l * P3, 512, 1, 0, 0, (int)P3, 0, 0, 0, 0, Bn);
}
